// Set2Set_2920577761285
// MI455X (gfx1250) — matrix operand order repaired and checked by reading, not compiled, not run on hardware
//
#include <hip/hip_runtime.h>

#define D   512        // IN_CHANNELS
#define D2  1024       // 2*D
#define D4  2048       // 4*D
#define NB  256        // BATCH_SIZE
#define NN  262144     // N_NODES
#define NSTEPS 6

typedef __attribute__((ext_vector_type(16))) __bf16 v16bf;
typedef __attribute__((ext_vector_type(8)))  float  v8f;

union FragBF { uint4 u[2]; v16bf v; };

// ---------------- utility kernels ----------------

__global__ void k_f32_to_bf16(const float* __restrict__ src, __bf16* __restrict__ dst, int n) {
  int i = blockIdx.x * blockDim.x + threadIdx.x;
  if (i < n) dst[i] = (__bf16)src[i];
}

__global__ void k_zero(float* __restrict__ p, int n) {
  int i = blockIdx.x * blockDim.x + threadIdx.x;
  if (i < n) p[i] = 0.0f;
}

__device__ __forceinline__ int lower_bound_i(const int* __restrict__ a, int n, int key) {
  int lo = 0, hi = n;
  while (lo < hi) { int mid = (lo + hi) >> 1; if (a[mid] < key) lo = mid + 1; else hi = mid; }
  return lo;
}

// segs[b] = first node of segment b, segs[NB+b] = one past last (batch is sorted)
__global__ void k_seg_bounds(const int* __restrict__ batch, int* __restrict__ segs) {
  int b = threadIdx.x;  // launched with 1 block of NB threads
  segs[b]      = lower_bound_i(batch, NN, b);
  segs[NB + b] = lower_bound_i(batch, NN, b + 1);
}

// ---------------- LSTM gates GEMM via WMMA (bf16 in, f32 accum) ----------------
// gates[b, n] = sum_k qs[b,k]*w_ih[n,k] + sum_k h[b,k]*w_hh[n,k] + b_ih[n] + b_hh[n]
// One 16x16 C tile per wave. 2048 tiles = 16 row-tiles x 128 col-tiles.
__global__ __launch_bounds__(256) void k_gates_wmma(
    const __bf16* __restrict__ qs_bf,   // [NB, D2]
    const __bf16* __restrict__ h_bf,    // [NB, D]
    const __bf16* __restrict__ wih_bf,  // [D4, D2] row-major
    const __bf16* __restrict__ whh_bf,  // [D4, D]  row-major
    const float* __restrict__ b_ih, const float* __restrict__ b_hh,
    float* __restrict__ gates)          // [NB, D4]
{
  const int wave = threadIdx.x >> 5;
  const int lane = threadIdx.x & 31;
  const int tile = blockIdx.x * 8 + wave;   // 0..2047
  const int tn   = tile & 127;              // column tile (N dim, 2048/16)
  const int tm   = tile >> 7;               // row tile    (M dim, 256/16)

  const int r  = lane & 15;
  const int hi = lane >> 4;                 // 0 or 1 (half-wave)
  const int aKoff = hi * 8;                 // A frag: K chunks {0..7,16..23} / {8..15,24..31}
  const int bKoff = hi * 8;                 // B frag: K chunks {0..7,16..23} / {8..15,24..31}

  const __bf16* __restrict__ arow = qs_bf  + (size_t)(tm * 16 + r) * D2;
  const __bf16* __restrict__ brow = wih_bf + (size_t)(tn * 16 + r) * D2;

  v8f acc = {};
  #pragma unroll 4
  for (int k = 0; k < D2; k += 32) {
    // pull the weight stream ahead (first touch: HBM->L2; later tiles hit L2)
    __builtin_prefetch(brow + k + 256, 0, 0);
    FragBF a, b;
    a.u[0] = *(const uint4*)(arow + k + aKoff);
    a.u[1] = *(const uint4*)(arow + k + 16 + aKoff);
    b.u[0] = *(const uint4*)(brow + k + bKoff);
    b.u[1] = *(const uint4*)(brow + k + 16 + bKoff);
    acc = __builtin_amdgcn_wmma_f32_16x16x32_bf16(false, a.v, false, b.v,
                                                  (short)0, acc, false, false);
  }

  const __bf16* __restrict__ arow2 = h_bf   + (size_t)(tm * 16 + r) * D;
  const __bf16* __restrict__ brow2 = whh_bf + (size_t)(tn * 16 + r) * D;
  #pragma unroll 4
  for (int k = 0; k < D; k += 32) {
    __builtin_prefetch(brow2 + k + 256, 0, 0);
    FragBF a, b;
    a.u[0] = *(const uint4*)(arow2 + k + aKoff);
    a.u[1] = *(const uint4*)(arow2 + k + 16 + aKoff);
    b.u[0] = *(const uint4*)(brow2 + k + bKoff);
    b.u[1] = *(const uint4*)(brow2 + k + 16 + bKoff);
    acc = __builtin_amdgcn_wmma_f32_16x16x32_bf16(false, a.v, false, b.v,
                                                  (short)0, acc, false, false);
  }

  const int n = tn * 16 + r;
  const float bias = b_ih[n] + b_hh[n];
  #pragma unroll
  for (int i = 0; i < 8; i++) {
    const int m = tm * 16 + i + hi * 8;     // C layout: VGPR i -> M = i (+8 for hi half)
    gates[(size_t)m * D4 + n] = acc[i] + bias;
  }
}

// ---------------- LSTM pointwise update ----------------
__global__ __launch_bounds__(256) void k_lstm(
    const float* __restrict__ gates, float* __restrict__ h,
    float* __restrict__ c, float* __restrict__ q_star)
{
  int idx = blockIdx.x * blockDim.x + threadIdx.x;  // 0 .. NB*D-1
  int b = idx >> 9, d = idx & (D - 1);
  const float* __restrict__ g = gates + (size_t)b * D4;
  float ig = 1.0f / (1.0f + expf(-g[d]));
  float fg = 1.0f / (1.0f + expf(-g[D + d]));
  float gg = tanhf(g[2 * D + d]);
  float og = 1.0f / (1.0f + expf(-g[3 * D + d]));
  float cc = fg * c[idx] + ig * gg;
  c[idx] = cc;
  float hh = og * tanhf(cc);
  h[idx] = hh;
  q_star[(size_t)b * D2 + d] = hh;   // left half of q_star = q = h
}

// ---------------- attention pass 1: e[n] = <x[n,:], h[batch[n],:]> ----------------
// one wave per node row; 16 f32 (4x float4) per lane; shuffle reduce
__global__ __launch_bounds__(256) void k_attn_dot(
    const float* __restrict__ x, const int* __restrict__ batch,
    const float* __restrict__ h, float* __restrict__ e)
{
  const int n    = blockIdx.x * 8 + (threadIdx.x >> 5);
  const int lane = threadIdx.x & 31;
  const float4* __restrict__ xr = (const float4*)(x + (size_t)n * D) + lane * 4;
  const float4* __restrict__ qr = (const float4*)(h + (size_t)batch[n] * D) + lane * 4;
  float s = 0.0f;
  #pragma unroll
  for (int i = 0; i < 4; i++) {
    float4 a = xr[i], b = qr[i];
    s = fmaf(a.x, b.x, s); s = fmaf(a.y, b.y, s);
    s = fmaf(a.z, b.z, s); s = fmaf(a.w, b.w, s);
  }
  #pragma unroll
  for (int off = 16; off > 0; off >>= 1) s += __shfl_xor(s, off, 32);
  if (lane == 0) e[n] = s;
}

// ---------------- per-segment softmax over e (in place -> a[n]) ----------------
__global__ __launch_bounds__(256) void k_attn_softmax(
    float* __restrict__ e, const int* __restrict__ segs)
{
  __shared__ float red[256];
  const int b = blockIdx.x, t = threadIdx.x;
  const int s0 = segs[b], s1 = segs[NB + b];

  float mx = -INFINITY;
  for (int n = s0 + t; n < s1; n += 256) mx = fmaxf(mx, e[n]);
  red[t] = mx; __syncthreads();
  #pragma unroll
  for (int w = 128; w > 0; w >>= 1) { if (t < w) red[t] = fmaxf(red[t], red[t + w]); __syncthreads(); }
  mx = red[0]; __syncthreads();

  float sum = 0.0f;
  for (int n = s0 + t; n < s1; n += 256) sum += expf(e[n] - mx);
  red[t] = sum; __syncthreads();
  #pragma unroll
  for (int w = 128; w > 0; w >>= 1) { if (t < w) red[t] += red[t + w]; __syncthreads(); }
  const float inv = 1.0f / (red[0] + 1e-16f);

  for (int n = s0 + t; n < s1; n += 256) e[n] = expf(e[n] - mx) * inv;
}

// ---------------- attention pass 2: q_star[b, D+col] = sum_n a[n] * x[n, col] ----------------
// grid (NB, 2): each block owns one segment and 256 columns; coalesced, no atomics.
// Access pattern per block: 1KB contiguous per row, 2KB row stride -> issue
// global_prefetch a few rows ahead to keep HBM busy under the fma chain.
__global__ __launch_bounds__(256) void k_attn_readout(
    const float* __restrict__ x, const float* __restrict__ a,
    const int* __restrict__ segs, float* __restrict__ q_star)
{
  const int b   = blockIdx.x;
  const int col = blockIdx.y * 256 + threadIdx.x;
  const int s0 = segs[b], s1 = segs[NB + b];
  float a0 = 0.f, a1 = 0.f, a2 = 0.f, a3 = 0.f;
  int n = s0;
  for (; n + 3 < s1; n += 4) {
    if (n + 16 < s1) __builtin_prefetch(x + (size_t)(n + 16) * D + col, 0, 0);
    a0 = fmaf(a[n + 0], x[(size_t)(n + 0) * D + col], a0);
    a1 = fmaf(a[n + 1], x[(size_t)(n + 1) * D + col], a1);
    a2 = fmaf(a[n + 2], x[(size_t)(n + 2) * D + col], a2);
    a3 = fmaf(a[n + 3], x[(size_t)(n + 3) * D + col], a3);
  }
  for (; n < s1; n++) a0 = fmaf(a[n], x[(size_t)n * D + col], a0);
  q_star[(size_t)b * D2 + D + col] = (a0 + a1) + (a2 + a3);
}

// ---------------- host launcher ----------------

extern "C" void kernel_launch(void* const* d_in, const int* in_sizes, int n_in,
                              void* d_out, int out_size, void* d_ws, size_t ws_size,
                              hipStream_t stream) {
  const float* x     = (const float*)d_in[0];
  const int*   batch = (const int*)d_in[1];
  const float* w_ih  = (const float*)d_in[2];   // [D4, D2]
  const float* w_hh  = (const float*)d_in[3];   // [D4, D]
  const float* b_ih  = (const float*)d_in[4];
  const float* b_hh  = (const float*)d_in[5];

  char* base = (char*)d_ws;
  size_t off = 0;
  auto carve = [&](size_t bytes) { char* p = base + off; off += (bytes + 255) & ~(size_t)255; return p; };

  float* h      = (float*)carve((size_t)NB * D  * 4);   // NOTE: h,c,q_star carved
  float* c      = (float*)carve((size_t)NB * D  * 4);   //       contiguously so one
  float* q_star = (float*)carve((size_t)NB * D2 * 4);   //       zero-fill covers all
  float* gates  = (float*)carve((size_t)NB * D4 * 4);
  float* e      = (float*)carve((size_t)NN * 4);
  int*   segs   = (int*)  carve((size_t)2 * NB * 4);
  __bf16* wih_bf = (__bf16*)carve((size_t)D4 * D2 * 2);
  __bf16* whh_bf = (__bf16*)carve((size_t)D4 * D  * 2);
  __bf16* qs_bf  = (__bf16*)carve((size_t)NB * D2 * 2);
  __bf16* h_bf   = (__bf16*)carve((size_t)NB * D  * 2);

  // one-time per launch: weight conversion, segment bounds, state init
  {
    int n1 = D4 * D2, n2 = D4 * D;
    k_f32_to_bf16<<<(n1 + 255) / 256, 256, 0, stream>>>(w_ih, wih_bf, n1);
    k_f32_to_bf16<<<(n2 + 255) / 256, 256, 0, stream>>>(w_hh, whh_bf, n2);
    k_seg_bounds<<<1, NB, 0, stream>>>(batch, segs);
    int nz = NB * D + NB * D + NB * D2;   // h + c + q_star (contiguous)
    k_zero<<<(nz + 255) / 256, 256, 0, stream>>>(h, nz);
  }

  for (int step = 0; step < NSTEPS; step++) {
    int nq = NB * D2, nh = NB * D;
    k_f32_to_bf16<<<(nq + 255) / 256, 256, 0, stream>>>(q_star, qs_bf, nq);
    k_f32_to_bf16<<<(nh + 255) / 256, 256, 0, stream>>>(h, h_bf, nh);

    k_gates_wmma<<<256, 256, 0, stream>>>(qs_bf, h_bf, wih_bf, whh_bf, b_ih, b_hh, gates);
    k_lstm<<<(NB * D) / 256, 256, 0, stream>>>(gates, h, c, q_star);

    k_attn_dot<<<NN / 8, 256, 0, stream>>>(x, batch, h, e);
    k_attn_softmax<<<NB, 256, 0, stream>>>(e, segs);
    k_attn_readout<<<dim3(NB, 2), 256, 0, stream>>>(x, e, segs, q_star);
  }

  hipMemcpyAsync(d_out, q_star, (size_t)NB * D2 * sizeof(float),
                 hipMemcpyDeviceToDevice, stream);
}
